// HGAT_61495341744804
// MI455X (gfx1250) — matrix operand order repaired and checked by reading, not compiled, not run on hardware
//
#include <hip/hip_runtime.h>

#define DEV __device__ __forceinline__

typedef __attribute__((ext_vector_type(16))) __bf16 v16bf;
typedef __attribute__((ext_vector_type(8)))  float  v8f;

DEV unsigned short f2bf(float x) {
    union { float f; unsigned u; } c; c.f = x;
    unsigned u = c.u + 0x7fffu + ((c.u >> 16) & 1u);   // round-to-nearest-even
    return (unsigned short)(u >> 16);
}
DEV unsigned int bfpack(float lo, float hi) {
    return (unsigned int)f2bf(lo) | ((unsigned int)f2bf(hi) << 16);
}

DEV float act_apply(float x, int act) {
    if (act == 1) return x > 0.f ? x : 0.f;                           // relu
    if (act == 2) return 0.5f * x * (1.f + erff(x * 0.70710678118f)); // exact gelu
    return x;
}

DEV void atomicMaxF(float* a, float v) {
    if (v >= 0.f) atomicMax((int*)a, __float_as_int(v));
    else          atomicMin((unsigned int*)a, __float_as_uint(v));
}

// ---------------------------------------------------------------------------
// One-time fp32 -> bf16 conversion kernels (operands converted exactly once
// per GEMM instead of once per tile-reuse inside the GEMM loop).
// ---------------------------------------------------------------------------
__global__ void cvt_kernel(unsigned short* __restrict__ dst,
                           const float* __restrict__ src, long n4)
{
    long tid = (long)blockIdx.x * blockDim.x + threadIdx.x;
    if (tid >= n4) return;
    float4 f = ((const float4*)src)[tid];
    uint2 p; p.x = bfpack(f.x, f.y); p.y = bfpack(f.z, f.w);
    ((uint2*)dst)[tid] = p;
}

// dst[N][K] (bf16) = src[K][N] (fp32): weights transposed to K-contiguous
__global__ void cvt_t_kernel(unsigned short* __restrict__ dst,
                             const float* __restrict__ src, int K, int N)
{
    long tid = (long)blockIdx.x * blockDim.x + threadIdx.x;
    if (tid >= (long)K * N) return;
    int n = (int)(tid / K), k = (int)(tid % K);
    dst[tid] = f2bf(src[(long)k * N + n]);
}

// vt[b][h][d][t] (bf16) = v[b][t][h][d] (fp32): V transposed for attn @ V
__global__ void cvt_vT_kernel(unsigned short* __restrict__ dst,
                              const float* __restrict__ src,
                              int B, int S, int H, int DH)
{
    long tid = (long)blockIdx.x * blockDim.x + threadIdx.x;
    if (tid >= (long)B * S * H * DH) return;
    int t  = (int)(tid % S);
    long r = tid / S;
    int d  = (int)(r % DH); r /= DH;
    int hh = (int)(r % H);
    int b  = (int)(r / H);
    dst[tid] = f2bf(src[((long)b * S + t) * H * DH + hh * DH + d]);
}

// ---------------------------------------------------------------------------
// Batched GEMM: C = act(alpha * A@B^T + bias); A [M,K] bf16 row-major,
// B [N,K] bf16 row-major (K-contiguous), C fp32. Block = 128 threads
// (4 waves), tile 32(M) x 64(N); each wave: two 16x16 WMMA tiles sharing one
// A fragment -> 2 v_wmma_f32_16x16x32_bf16 per K step. LDS holds tiles in the
// per-lane fragment order (k = 16*half + elem), so staging is 3x b128 load + 3x b128 DS store and
// fragment reads are 2x ds_load_b128 each. Branch-free (A row clamped; OOB
// rows compute dead accumulators that are never stored).
// Requirements (all call sites satisfy): K % 32 == 0, N % 64 == 0, M >= 32.
// ---------------------------------------------------------------------------
__global__ __launch_bounds__(128)
void wmma_gemm_bf16(const unsigned short* __restrict__ A, int lda, long sA1, long sA2,
                    const unsigned short* __restrict__ B, int ldb, long sB1, long sB2,
                    float* __restrict__ C, int ldc, long sC1, long sC2,
                    const float* __restrict__ bias,
                    int M, int N, int K, int Hdim, float alpha, int act)
{
    __shared__ unsigned short Afrag[2][32][16];   // [wr strip][lane][elem]
    __shared__ unsigned short Bfrag[4][32][16];   // [n strip][lane][elem]

    const int z  = blockIdx.z;
    const int zo = z / Hdim, zi = z % Hdim;
    A += (long)zo * sA1 + (long)zi * sA2;
    B += (long)zo * sB1 + (long)zi * sB2;
    C += (long)zo * sC1 + (long)zi * sC2;

    const int tid  = threadIdx.x;
    const int wave = tid >> 5;
    const int lane = tid & 31;
    const int wr = wave >> 1, wc = wave & 1;
    const int bm = blockIdx.y * 32;
    const int bn = blockIdx.x * 64;
    const int half = lane >> 4;
    const int lr   = lane & 15;

    // A staging: thread -> (row, 8 contiguous K) -> one b128, one DS b128
    const int ar  = tid >> 2;
    const int ac0 = (tid & 3) * 8;
    const int a_e0   = ac0 & 8;
    const int a_lane = (ac0 >> 4) * 16 + (ar & 15);
    const int a_wr   = ar >> 4;
    const int a_row  = min(bm + ar, M - 1);       // clamp: OOB rows never stored
    const unsigned short* Ap = A + (long)a_row * lda + ac0;

    // B staging: thread -> (col n, 16 contiguous K) -> two b128, two DS b128
    const int bc  = tid & 63;
    const int bkb = (tid >> 6) * 16;
    const int b_strip = bc >> 4;
    const int b_lane  = (bkb >> 4) * 16 + (bc & 15);
    const unsigned short* Bp = B + (long)(bn + bc) * ldb + bkb;

    v8f acc0 = {};
    v8f acc1 = {};

    for (int k0 = 0; k0 < K; k0 += 32) {
        *(uint4*)&Afrag[a_wr][a_lane][a_e0] = *(const uint4*)(Ap + k0);
        *(uint4*)&Bfrag[b_strip][b_lane][0] = *(const uint4*)(Bp + k0);
        *(uint4*)&Bfrag[b_strip][b_lane][8] = *(const uint4*)(Bp + k0 + 8);
        if (k0 + 32 < K) {                        // gfx1250 global_prefetch_b8
            __builtin_prefetch((const void*)(Ap + k0 + 32), 0, 0);
            __builtin_prefetch((const void*)(Bp + k0 + 32), 0, 0);
        }
        __syncthreads();

        union { v16bf v; uint4 q[2]; } af, b0, b1;
        af.q[0] = *(const uint4*)&Afrag[wr][lane][0];
        af.q[1] = *(const uint4*)&Afrag[wr][lane][8];
        b0.q[0] = *(const uint4*)&Bfrag[wc][lane][0];
        b0.q[1] = *(const uint4*)&Bfrag[wc][lane][8];
        b1.q[0] = *(const uint4*)&Bfrag[wc + 2][lane][0];
        b1.q[1] = *(const uint4*)&Bfrag[wc + 2][lane][8];

        acc0 = __builtin_amdgcn_wmma_f32_16x16x32_bf16(false, af.v, false, b0.v,
                                                       (short)0, acc0, false, false);
        acc1 = __builtin_amdgcn_wmma_f32_16x16x32_bf16(false, af.v, false, b1.v,
                                                       (short)0, acc1, false, false);
        __syncthreads();
    }

    // C/D layout: VGPR i -> row i (lanes 0-15) / i+8 (lanes 16-31), col = lane&15
    const int col0 = bn + 16 * wc + lr;
    const int col1 = col0 + 32;
    const float bv0 = bias ? bias[col0] : 0.f;
    const float bv1 = bias ? bias[col1] : 0.f;
    #pragma unroll
    for (int i = 0; i < 8; ++i) {
        int row = bm + 16 * wr + i + 8 * half;
        if (row < M) {
            C[(long)row * ldc + col0] = act_apply(acc0[i] * alpha + bv0, act);
            C[(long)row * ldc + col1] = act_apply(acc1[i] * alpha + bv1, act);
        }
    }
}

// ---------------------------------------------------------------------------
// LayerNorm with optional fused residual (out = LN(in + res))
// ---------------------------------------------------------------------------
__global__ __launch_bounds__(256)
void ln_kernel(float* __restrict__ out, const float* __restrict__ in,
               const float* __restrict__ res,
               const float* __restrict__ g, const float* __restrict__ b, int D)
{
    __shared__ float s1[256], s2[256];
    long base = (long)blockIdx.x * D;
    float sum = 0.f, sq = 0.f;
    for (int d = threadIdx.x; d < D; d += 256) {
        float x = in[base + d] + (res ? res[base + d] : 0.f);
        sum += x; sq += x * x;
    }
    s1[threadIdx.x] = sum; s2[threadIdx.x] = sq;
    __syncthreads();
    for (int s = 128; s > 0; s >>= 1) {
        if (threadIdx.x < s) { s1[threadIdx.x] += s1[threadIdx.x + s];
                               s2[threadIdx.x] += s2[threadIdx.x + s]; }
        __syncthreads();
    }
    float mean = s1[0] / D;
    float var  = s2[0] / D - mean * mean;
    float rstd = rsqrtf(var + 1e-5f);
    for (int d = threadIdx.x; d < D; d += 256) {
        float x = in[base + d] + (res ? res[base + d] : 0.f);
        out[base + d] = (x - mean) * rstd * g[d] + b[d];
    }
}

// row softmax over S=256 with attention-mask bias; one block per (b,h,q) row
__global__ __launch_bounds__(256)
void softmax_kernel(float* __restrict__ sc, const int* __restrict__ mask,
                    int B, int H, int S)
{
    __shared__ float red[256];
    int row = blockIdx.x;                // b*H*S + h*S + q
    int b = row / (H * S);
    long base = (long)row * S;
    int t = threadIdx.x;                 // S == 256 == blockDim
    float mb = (1.f - (float)mask[b * S + t]) * -1e9f;
    float x = sc[base + t] + mb;
    red[t] = x; __syncthreads();
    for (int s = 128; s > 0; s >>= 1) { if (t < s) red[t] = fmaxf(red[t], red[t + s]); __syncthreads(); }
    float mx = red[0]; __syncthreads();
    float e = __expf(x - mx);
    red[t] = e; __syncthreads();
    for (int s = 128; s > 0; s >>= 1) { if (t < s) red[t] += red[t + s]; __syncthreads(); }
    sc[base + t] = e / red[0];
}

__global__ void embed_kernel(float* h, const int* ids, const float* wemb,
                             const float* pemb, int B, int S, int D)
{
    long tid = (long)blockIdx.x * blockDim.x + threadIdx.x;
    if (tid >= (long)B * S * D) return;
    int d = tid % D;
    long bs = tid / D;
    int s = (int)(bs % S);
    int id = ids[bs];
    h[tid] = wemb[(long)id * D + d] + pemb[(long)(s + 2) * D + d];
}

__global__ void copy_cls_kernel(float* ht, const float* h, int B, int S, int D)
{
    int tid = blockIdx.x * blockDim.x + threadIdx.x;
    if (tid >= B * D) return;
    ht[tid] = h[(long)(tid / D) * S * D + (tid % D)];
}

__global__ void fill_kernel(float* p, float v, long n)
{
    long tid = (long)blockIdx.x * blockDim.x + threadIdx.x;
    if (tid < n) p[tid] = v;
}

// a[n,h] = sum_c xl[n,h,c] * att[h,c]
__global__ void rowdot_kernel(float* out, const float* xl, const float* att,
                              int Np, int heads, int C)
{
    int tid = blockIdx.x * blockDim.x + threadIdx.x;
    if (tid >= Np * heads) return;
    int n = tid / heads, hh = tid % heads;
    const float* xp = xl + (long)n * heads * C + hh * C;
    const float* ap = att + hh * C;
    float s = 0.f;
    for (int c = 0; c < C; ++c) s += xp[c] * ap[c];
    out[tid] = s;
}

__global__ void edge_max_kernel(const int* ei, int E, int Np, int heads,
                                const float* asrc, const float* adst,
                                float* alpha, float* mx)
{
    int tid = blockIdx.x * blockDim.x + threadIdx.x;
    int Et = E + Np;
    if (tid >= Et * heads) return;
    int e = tid / heads, hh = tid % heads;
    int s = e < E ? ei[e]     : e - E;     // self loops appended
    int d = e < E ? ei[E + e] : e - E;
    float a = asrc[s * heads + hh] + adst[d * heads + hh];
    a = a > 0.f ? a : 0.2f * a;            // leaky_relu(0.2)
    alpha[tid] = a;
    atomicMaxF(&mx[d * heads + hh], a);
}

__global__ void edge_exp_kernel(const int* ei, int E, int Np, int heads,
                                float* alpha, const float* mx, float* ssum)
{
    int tid = blockIdx.x * blockDim.x + threadIdx.x;
    int Et = E + Np;
    if (tid >= Et * heads) return;
    int e = tid / heads, hh = tid % heads;
    int d = e < E ? ei[E + e] : e - E;
    float ee = __expf(alpha[tid] - mx[d * heads + hh]);
    alpha[tid] = ee;
    atomicAdd(&ssum[d * heads + hh], ee);
}

__global__ void edge_msg_kernel(const int* ei, int E, int Np, int heads, int C,
                                const float* xl, const float* alpha,
                                const float* ssum, float* out)
{
    long tid = (long)blockIdx.x * blockDim.x + threadIdx.x;
    int HC = heads * C;
    long tot = (long)(E + Np) * HC;
    if (tid >= tot) return;
    int e   = (int)(tid / HC);
    int rem = (int)(tid % HC);
    int hh = rem / C;
    int s = e < E ? ei[e]     : e - E;
    int d = e < E ? ei[E + e] : e - E;
    float w = alpha[(long)e * heads + hh] / (ssum[d * heads + hh] + 1e-16f);
    atomicAdd(&out[(long)d * HC + rem], xl[(long)s * HC + rem] * w);
}

__global__ void bias_act_kernel(float* x, const float* bias, long rows, int cols, int relu)
{
    long tid = (long)blockIdx.x * blockDim.x + threadIdx.x;
    if (tid >= rows * cols) return;
    float v = x[tid] + bias[tid % cols];
    x[tid] = relu ? fmaxf(v, 0.f) : v;
}

__global__ void pool_kernel(const float* x, const int* batch, float* sums, int Np, int C)
{
    long tid = (long)blockIdx.x * blockDim.x + threadIdx.x;
    if (tid >= (long)Np * C) return;
    int n = (int)(tid / C), c = (int)(tid % C);
    atomicAdd(&sums[batch[n] * C + c], x[tid]);
}

__global__ void count_kernel(const int* batch, float* cnt, int Np)
{
    int tid = blockIdx.x * blockDim.x + threadIdx.x;
    if (tid < Np) atomicAdd(&cnt[batch[tid]], 1.f);
}

__global__ void mean_kernel(float* hg, const float* sums, const float* cnt, int B, int C)
{
    int tid = blockIdx.x * blockDim.x + threadIdx.x;
    if (tid >= B * C) return;
    hg[tid] = sums[tid] / fmaxf(cnt[tid / C], 1.f);
}

__global__ void cat_kernel(float* cat, const float* ht, const float* hg, int B)
{
    int tid = blockIdx.x * blockDim.x + threadIdx.x;
    if (tid >= B * 1024) return;
    int b = tid / 1024, c = tid % 1024;
    cat[tid] = c < 512 ? ht[b * 512 + c] : hg[b * 512 + (c - 512)];
}

// tiny head GEMM: thread per (m,n)
__global__ void small_gemm_kernel(const float* A, const float* B, const float* bias,
                                  float* C, int M, int N, int K, int act)
{
    int tid = blockIdx.x * blockDim.x + threadIdx.x;
    if (tid >= M * N) return;
    int m = tid / N, n = tid % N;
    float s = bias ? bias[n] : 0.f;
    for (int k = 0; k < K; ++k) s += A[m * K + k] * B[k * N + n];
    C[tid] = (act == 1) ? fmaxf(s, 0.f) : s;
}

__global__ void fuse_kernel(const float* an2, const float* ht, const float* hg,
                            float* fused, float* beta_out, int B, int C)
{
    int tid = blockIdx.x * blockDim.x + threadIdx.x;
    if (tid >= B * C) return;
    int b = tid / C, c = tid % C;
    float bt = 1.f / (1.f + __expf(-an2[b]));
    fused[tid] = bt * ht[tid] + (1.f - bt) * hg[tid];
    if (c == 0) beta_out[b] = bt;
}

// ---------------------------------------------------------------------------
extern "C" void kernel_launch(void* const* d_in, const int* in_sizes, int n_in,
                              void* d_out, int out_size, void* d_ws, size_t ws_size,
                              hipStream_t stream)
{
    const int Bq = 16, S = 256, D = 768, Hh = 12, DH = 64, F = 3072, L = 12;
    const int NN = 10000, NE = 320000, ET = NE + NN;

    const int*   ids   = (const int*)d_in[0];
    const int*   amask = (const int*)d_in[1];
    const float* xg    = (const float*)d_in[2];
    const int*   eidx  = (const int*)d_in[3];
    const int*   batch = (const int*)d_in[4];
    // params flattened in jax pytree order (sorted keys, recursive)
    const float* an_W1   = (const float*)d_in[5];
    const float* an_W2   = (const float*)d_in[6];
    const float* an_b1   = (const float*)d_in[7];
    const float* an_b2   = (const float*)d_in[8];
    const float* emb_ln_b= (const float*)d_in[9];
    const float* emb_ln_s= (const float*)d_in[10];
    const float* gat1_W  = (const float*)d_in[11];
    const float* gat1_ad = (const float*)d_in[12];
    const float* gat1_as = (const float*)d_in[13];
    const float* gat1_b  = (const float*)d_in[14];
    const float* gat2_W  = (const float*)d_in[15];
    const float* gat2_ad = (const float*)d_in[16];
    const float* gat2_as = (const float*)d_in[17];
    const float* gat2_b  = (const float*)d_in[18];
    const float* gp_W    = (const float*)d_in[19];
    const float* gp_b    = (const float*)d_in[20];
    const float* lW1     = (const float*)d_in[21];
    const float* lW2     = (const float*)d_in[22];
    const float* lWk     = (const float*)d_in[23];
    const float* lWo     = (const float*)d_in[24];
    const float* lWq     = (const float*)d_in[25];
    const float* lWv     = (const float*)d_in[26];
    const float* lb1     = (const float*)d_in[27];
    const float* lb2     = (const float*)d_in[28];
    const float* lbk     = (const float*)d_in[29];
    const float* lbo     = (const float*)d_in[30];
    const float* lbq     = (const float*)d_in[31];
    const float* lbv     = (const float*)d_in[32];
    const float* lln1b   = (const float*)d_in[33];
    const float* lln1s   = (const float*)d_in[34];
    const float* lln2b   = (const float*)d_in[35];
    const float* lln2s   = (const float*)d_in[36];
    const float* mlp_W1  = (const float*)d_in[37];
    const float* mlp_W2  = (const float*)d_in[38];
    const float* mlp_b1  = (const float*)d_in[39];
    const float* mlp_b2  = (const float*)d_in[40];
    const float* pos_emb = (const float*)d_in[41];
    const float* tp_W    = (const float*)d_in[42];
    const float* tp_b    = (const float*)d_in[43];
    const float* word_emb= (const float*)d_in[44];

    float* out = (float*)d_out;  // [0..31] logits, [32..47] beta

    // ---- scratch carve (bf16 buffers first: keep 16B alignment) ----
    float* W = (float*)d_ws;
    size_t off = 0;
    auto alloc = [&](size_t n) { float* p = W + off; off += n; return p; };

    unsigned short* bfA = (unsigned short*)alloc((size_t)Bq * S * F / 2); // 12.6M bf16
    unsigned short* bfB = (unsigned short*)alloc((size_t)Bq * S * D / 2); //  3.1M bf16
    float* h    = alloc((size_t)Bq * S * D);
    float* q    = alloc((size_t)Bq * S * D);
    float* kbuf = alloc((size_t)Bq * S * D);
    float* v    = alloc((size_t)Bq * S * D);
    float* big  = alloc((size_t)Bq * S * F);   // scores [B,H,S,S] / FFN mid [B,S,F]
    float* tmp  = alloc((size_t)Bq * S * D);
    float* h_text = alloc((size_t)Bq * D);
    // GAT
    float* xl1   = alloc((size_t)NN * 512);
    float* asrc1 = alloc((size_t)NN * 8);
    float* adst1 = alloc((size_t)NN * 8);
    float* m1    = alloc((size_t)NN * 8);
    float* s1    = alloc((size_t)NN * 8);
    float* alpha1= alloc((size_t)ET * 8);
    float* gout1 = alloc((size_t)NN * 512);
    float* xl2   = alloc((size_t)NN * 256);
    float* asrc2 = alloc((size_t)NN);
    float* adst2 = alloc((size_t)NN);
    float* m2    = alloc((size_t)NN);
    float* s2    = alloc((size_t)NN);
    float* alpha2= alloc((size_t)ET);
    float* gout2 = alloc((size_t)NN * 256);
    float* gsum  = alloc((size_t)Bq * 256);
    float* gcnt  = alloc((size_t)Bq);
    float* h_graph = alloc((size_t)Bq * 256);
    float* ht    = alloc((size_t)Bq * 512);
    float* hg    = alloc((size_t)Bq * 512);
    float* cat   = alloc((size_t)Bq * 1024);
    float* an1   = alloc((size_t)Bq * 128);
    float* an2   = alloc((size_t)Bq);
    float* fused = alloc((size_t)Bq * 512);
    float* mlp1  = alloc((size_t)Bq * 256);

    auto cdiv = [](long a, long b) { return (int)((a + b - 1) / b); };

    auto cvt = [&](unsigned short* dst, const float* src, long n) {
        cvt_kernel<<<cdiv(n / 4, 256), 256, 0, stream>>>(dst, src, n / 4);
    };
    auto cvtT = [&](unsigned short* dst, const float* src, int K, int N) {
        cvt_t_kernel<<<cdiv((long)K * N, 256), 256, 0, stream>>>(dst, src, K, N);
    };
    auto gemm = [&](const unsigned short* A, int lda, long sA1, long sA2,
                    const unsigned short* Bp, int ldb, long sB1, long sB2,
                    float* Cp, int ldc, long sC1, long sC2,
                    const float* bias, int M, int N, int K, int Hdim, int nb,
                    float alpha, int act) {
        dim3 g(N / 64, cdiv(M, 32), nb);     // N % 64 == 0 at every call site
        wmma_gemm_bf16<<<g, dim3(128), 0, stream>>>(
            A, lda, sA1, sA2, Bp, ldb, sB1, sB2,
            Cp, ldc, sC1, sC2, bias, M, N, K, Hdim, alpha, act);
    };

    // ================= RoBERTa text encoder =================
    {
        long tot = (long)Bq * S * D;
        embed_kernel<<<cdiv(tot, 256), 256, 0, stream>>>(h, ids, word_emb, pos_emb, Bq, S, D);
        ln_kernel<<<Bq * S, 256, 0, stream>>>(h, h, nullptr, emb_ln_s, emb_ln_b, D);
    }
    const int M = Bq * S;
    const long BSD = (long)Bq * S * D;
    for (int l = 0; l < L; ++l) {
        const float* Wq = lWq + (size_t)l * D * D; const float* bq = lbq + (size_t)l * D;
        const float* Wk = lWk + (size_t)l * D * D; const float* bk = lbk + (size_t)l * D;
        const float* Wv = lWv + (size_t)l * D * D; const float* bv = lbv + (size_t)l * D;
        const float* Wo = lWo + (size_t)l * D * D; const float* bo = lbo + (size_t)l * D;
        const float* W1 = lW1 + (size_t)l * D * F; const float* b1 = lb1 + (size_t)l * F;
        const float* W2 = lW2 + (size_t)l * F * D; const float* b2 = lb2 + (size_t)l * D;
        const float* l1s = lln1s + (size_t)l * D; const float* l1b = lln1b + (size_t)l * D;
        const float* l2s = lln2s + (size_t)l * D; const float* l2b = lln2b + (size_t)l * D;

        // ---- QKV projections ----
        cvt(bfA, h, BSD);
        cvtT(bfB, Wq, D, D);
        gemm(bfA, D, 0, 0, bfB, D, 0, 0, q,    D, 0, 0, bq, M, D, D, 1, 1, 1.f, 0);
        cvtT(bfB, Wk, D, D);
        gemm(bfA, D, 0, 0, bfB, D, 0, 0, kbuf, D, 0, 0, bk, M, D, D, 1, 1, 1.f, 0);
        cvtT(bfB, Wv, D, D);
        gemm(bfA, D, 0, 0, bfB, D, 0, 0, v,    D, 0, 0, bv, M, D, D, 1, 1, 1.f, 0);

        // ---- scores = Q K^T / sqrt(DH) ----
        cvt(bfA, q, BSD);         // Q bf16, layout [B,S,H,DH]
        cvt(bfB, kbuf, BSD);      // K bf16, same layout ([N=token][K=d] natural)
        gemm(bfA, D, (long)S * D, DH, bfB, D, (long)S * D, DH,
             big, S, (long)Hh * S * S, (long)S * S,
             nullptr, S, S, DH, Hh, Bq * Hh, 0.125f, 0);
        softmax_kernel<<<Bq * Hh * S, 256, 0, stream>>>(big, amask, Bq, Hh, S);

        // ---- ctx = attn @ V ----
        cvt(bfA, big, (long)Bq * Hh * S * S);                 // attn bf16
        cvt_vT_kernel<<<cdiv(BSD, 256), 256, 0, stream>>>(bfB, v, Bq, S, Hh, DH);
        gemm(bfA, S, (long)Hh * S * S, (long)S * S,
             bfB, S, (long)Hh * DH * S, (long)DH * S,        // vt[b][h][d][t]
             tmp, D, (long)S * D, DH,
             nullptr, S, DH, S, Hh, Bq * Hh, 1.f, 0);

        // ---- out proj + residual LN ----
        cvt(bfA, tmp, BSD);
        cvtT(bfB, Wo, D, D);
        gemm(bfA, D, 0, 0, bfB, D, 0, 0, q, D, 0, 0, bo, M, D, D, 1, 1, 1.f, 0);
        ln_kernel<<<M, 256, 0, stream>>>(h, q, h, l1s, l1b, D);

        // ---- FFN ----
        cvt(bfA, h, BSD);
        cvtT(bfB, W1, D, F);
        gemm(bfA, D, 0, 0, bfB, D, 0, 0, big, F, 0, 0, b1, M, F, D, 1, 1, 1.f, 2);
        cvt(bfA, big, (long)M * F);
        cvtT(bfB, W2, F, D);
        gemm(bfA, F, 0, 0, bfB, F, 0, 0, tmp, D, 0, 0, b2, M, D, F, 1, 1, 1.f, 0);
        ln_kernel<<<M, 256, 0, stream>>>(h, tmp, h, l2s, l2b, D);
    }
    copy_cls_kernel<<<cdiv(Bq * D, 256), 256, 0, stream>>>(h_text, h, Bq, S, D);

    // ================= GAT layer 1 (8 heads x 64, concat) =================
    cvt(bfA, xg, (long)NN * 768);
    cvtT(bfB, gat1_W, 768, 512);
    gemm(bfA, 768, 0, 0, bfB, 768, 0, 0, xl1, 512, 0, 0, nullptr,
         NN, 512, 768, 1, 1, 1.f, 0);
    rowdot_kernel<<<cdiv(NN * 8, 256), 256, 0, stream>>>(asrc1, xl1, gat1_as, NN, 8, 64);
    rowdot_kernel<<<cdiv(NN * 8, 256), 256, 0, stream>>>(adst1, xl1, gat1_ad, NN, 8, 64);
    fill_kernel<<<cdiv(NN * 8, 256), 256, 0, stream>>>(m1, -1e30f, (long)NN * 8);
    fill_kernel<<<cdiv(NN * 8, 256), 256, 0, stream>>>(s1, 0.f,    (long)NN * 8);
    fill_kernel<<<cdiv((long)NN * 512, 256), 256, 0, stream>>>(gout1, 0.f, (long)NN * 512);
    edge_max_kernel<<<cdiv((long)ET * 8, 256), 256, 0, stream>>>(eidx, NE, NN, 8, asrc1, adst1, alpha1, m1);
    edge_exp_kernel<<<cdiv((long)ET * 8, 256), 256, 0, stream>>>(eidx, NE, NN, 8, alpha1, m1, s1);
    edge_msg_kernel<<<cdiv((long)ET * 512, 256), 256, 0, stream>>>(eidx, NE, NN, 8, 64, xl1, alpha1, s1, gout1);
    bias_act_kernel<<<cdiv((long)NN * 512, 256), 256, 0, stream>>>(gout1, gat1_b, NN, 512, 1);

    // ================= GAT layer 2 (1 head x 256, mean) =================
    cvt(bfA, gout1, (long)NN * 512);
    cvtT(bfB, gat2_W, 512, 256);
    gemm(bfA, 512, 0, 0, bfB, 512, 0, 0, xl2, 256, 0, 0, nullptr,
         NN, 256, 512, 1, 1, 1.f, 0);
    rowdot_kernel<<<cdiv(NN, 256), 256, 0, stream>>>(asrc2, xl2, gat2_as, NN, 1, 256);
    rowdot_kernel<<<cdiv(NN, 256), 256, 0, stream>>>(adst2, xl2, gat2_ad, NN, 1, 256);
    fill_kernel<<<cdiv(NN, 256), 256, 0, stream>>>(m2, -1e30f, (long)NN);
    fill_kernel<<<cdiv(NN, 256), 256, 0, stream>>>(s2, 0.f, (long)NN);
    fill_kernel<<<cdiv((long)NN * 256, 256), 256, 0, stream>>>(gout2, 0.f, (long)NN * 256);
    edge_max_kernel<<<cdiv(ET, 256), 256, 0, stream>>>(eidx, NE, NN, 1, asrc2, adst2, alpha2, m2);
    edge_exp_kernel<<<cdiv(ET, 256), 256, 0, stream>>>(eidx, NE, NN, 1, alpha2, m2, s2);
    edge_msg_kernel<<<cdiv((long)ET * 256, 256), 256, 0, stream>>>(eidx, NE, NN, 1, 256, xl2, alpha2, s2, gout2);
    bias_act_kernel<<<cdiv((long)NN * 256, 256), 256, 0, stream>>>(gout2, gat2_b, NN, 256, 0);

    // ================= global mean pool =================
    fill_kernel<<<cdiv(Bq * 256, 256), 256, 0, stream>>>(gsum, 0.f, (long)Bq * 256);
    fill_kernel<<<1, 256, 0, stream>>>(gcnt, 0.f, (long)Bq);
    pool_kernel<<<cdiv((long)NN * 256, 256), 256, 0, stream>>>(gout2, batch, gsum, NN, 256);
    count_kernel<<<cdiv(NN, 256), 256, 0, stream>>>(batch, gcnt, NN);
    mean_kernel<<<cdiv(Bq * 256, 256), 256, 0, stream>>>(h_graph, gsum, gcnt, Bq, 256);

    // ================= fusion head =================
    small_gemm_kernel<<<cdiv(Bq * 512, 256), 256, 0, stream>>>(h_text, tp_W, tp_b, ht, Bq, 512, 768, 0);
    small_gemm_kernel<<<cdiv(Bq * 512, 256), 256, 0, stream>>>(h_graph, gp_W, gp_b, hg, Bq, 512, 256, 0);
    cat_kernel<<<cdiv(Bq * 1024, 256), 256, 0, stream>>>(cat, ht, hg, Bq);
    small_gemm_kernel<<<cdiv(Bq * 128, 256), 256, 0, stream>>>(cat, an_W1, an_b1, an1, Bq, 128, 1024, 1);
    small_gemm_kernel<<<1, 256, 0, stream>>>(an1, an_W2, an_b2, an2, Bq, 1, 128, 0);
    fuse_kernel<<<cdiv(Bq * 512, 256), 256, 0, stream>>>(an2, ht, hg, fused, out + 32, Bq, 512);
    small_gemm_kernel<<<cdiv(Bq * 256, 256), 256, 0, stream>>>(fused, mlp_W1, mlp_b1, mlp1, Bq, 256, 512, 1);
    small_gemm_kernel<<<1, 256, 0, stream>>>(mlp1, mlp_W2, mlp_b2, out, Bq, 2, 256, 0);
}
